// CBraMod_57612691309057
// MI455X (gfx1250) — matrix operand order repaired and checked by reading, not compiled, not run on hardware
//
#include <hip/hip_runtime.h>
#include <hip/hip_bf16.h>
#include <math.h>

// ---------------------------------------------------------------------------
// Types for CDNA5 WMMA (wave32, 16x16x32 bf16 -> f32)
// ---------------------------------------------------------------------------
typedef __bf16 bf16_t;
typedef __bf16  v16bf __attribute__((ext_vector_type(16)));
typedef __bf16  v8bf  __attribute__((ext_vector_type(8)));
typedef float   v8f   __attribute__((ext_vector_type(8)));
typedef int     v4i_vs __attribute__((vector_size(16)));              // async-builtin param 0
typedef __attribute__((address_space(3))) v4i_vs* lds_v4i_p;          // async-builtin param 1

// CDNA5 async global->LDS copy (ASYNCcnt path), with safe fallback
#if defined(__has_builtin)
# if __has_builtin(__builtin_amdgcn_global_load_async_to_lds_b128) && \
     __has_builtin(__builtin_amdgcn_s_wait_asynccnt)
#  define GEMM_ASYNC 1
# endif
#endif
#ifndef GEMM_ASYNC
# define GEMM_ASYNC 0
#endif

// ---------------------------------------------------------------------------
// Problem constants
// ---------------------------------------------------------------------------
constexpr int BB   = 256;          // batch
constexpr int CC   = 19;           // channels
constexpr int PP   = 30;           // patches
constexpr int TL   = 200;          // time
constexpr int DD   = 200;          // model dim
constexpr int DFF  = 800;          // expert hidden
constexpr int NE   = 4;            // experts
constexpr int OC   = 25;           // conv channels
constexpr int SS   = CC * PP;      // 570 spatial rows for temporal convs
constexpr int TP   = 8;            // temporal length after stride-25 conv
constexpr int TT   = BB * CC * PP; // 145920 tokens
constexpr int CAP  = TT / NE;      // 36480 capacity
constexpr int KD   = 224;          // d padded to mult of 32 (K for FFN1)
constexpr int KF   = 128;          // 101 freq padded to mult of 32
constexpr int NP   = 224;          // 200 padded to mult of 32 (B-matrix rows)
constexpr int NFREQ = 101;

constexpr size_t alup(size_t x) { return (x + 255) & ~size_t(255); }

// workspace layout (bytes)
constexpr size_t SZ_H      = (size_t)BB * OC * SS * TP * 4;      // 116,736,000
constexpr size_t OFF_HA    = 0;
constexpr size_t OFF_HB    = alup(OFF_HA + SZ_H);
constexpr size_t OFF_PEMB  = alup(OFF_HB + SZ_H);                // f32 TT x D
constexpr size_t OFF_AMP   = alup(OFF_PEMB + (size_t)TT * DD * 4);   // bf16 TT x KF
constexpr size_t OFF_TW    = alup(OFF_AMP + (size_t)TT * KF * 2);    // cos+sin 101x200 f32
constexpr size_t OFF_WSPEC = alup(OFF_TW + 2ull * NFREQ * TL * 4);   // bf16 NP x KF
constexpr size_t OFF_W1T   = alup(OFF_WSPEC + (size_t)NP * KF * 2);  // bf16 NE x DFF x KD
constexpr size_t OFF_W2T   = alup(OFF_W1T + (size_t)NE * DFF * KD * 2); // bf16 NE x NP x DFF
constexpr size_t OFF_EXP   = alup(OFF_W2T + (size_t)NE * NP * DFF * 2);
constexpr size_t OFF_GATE  = alup(OFF_EXP + (size_t)TT * 4);
constexpr size_t OFF_LRNK  = alup(OFF_GATE + (size_t)TT * 4);
constexpr size_t OFF_BCNT  = alup(OFF_LRNK + (size_t)TT * 4);    // 570*4 ints
constexpr size_t OFF_BBAS  = alup(OFF_BCNT + 570ull * 4 * 4);
constexpr size_t OFF_IDX   = alup(OFF_BBAS + 570ull * 4 * 4);
constexpr size_t OFF_GK    = alup(OFF_IDX + (size_t)TT * 4);
constexpr size_t OFF_HID   = alup(OFF_GK + (size_t)TT * 4);      // bf16 NE x CAP x DFF
// reuse: buf (bf16 TT x KD = 65MB) overlays dead patch_emb; y (f32 TT x D) overlays hA; pe overlays hB
constexpr size_t OFF_BUF = OFF_PEMB;
constexpr size_t OFF_Y   = OFF_HA;
constexpr size_t OFF_PE  = OFF_HB;

// ---------------------------------------------------------------------------
// conv1: [B,1,570,200] * [25,1,1,49], stride 25, pad 24  -> h[B,25,570,8]
// ---------------------------------------------------------------------------
__global__ __launch_bounds__(256) void k_conv1(const float* __restrict__ x,
                                               const float* __restrict__ w,
                                               const float* __restrict__ bias,
                                               float* __restrict__ h) {
    long i = (long)blockIdx.x * 256 + threadIdx.x;
    if (i >= (long)BB * OC * SS * TP) return;
    int tp = i & 7;
    long r  = i >> 3;
    int s  = r % SS; long r2 = r / SS;
    int oc = r2 % OC; int b = r2 / OC;
    const float* xr = x + ((size_t)b * SS + s) * TL;
    const float* wr = w + oc * 49;
    float acc = bias[oc];
    int t0 = tp * 25 - 24;
#pragma unroll 7
    for (int k = 0; k < 49; k++) {
        int t = t0 + k;
        if (t >= 0 && t < TL) acc += wr[k] * xr[t];
    }
    h[i] = acc;
}

// ---------------------------------------------------------------------------
// GroupNorm(5 groups) + exact GELU, in place over h[B,25,570,8]
// one block per (b, group); group data is contiguous 5*570*8 floats
// ---------------------------------------------------------------------------
__global__ __launch_bounds__(256) void k_gn_gelu(float* __restrict__ h,
                                                 const float* __restrict__ gs,
                                                 const float* __restrict__ gb) {
    int b = blockIdx.x / 5, g = blockIdx.x % 5;
    int tid = threadIdx.x;
    float* base = h + (((size_t)b * OC + g * 5) * SS) * TP;
    const int n = 5 * SS * TP; // 22800
    float s = 0.f, s2 = 0.f;
    for (int i = tid; i < n; i += 256) { float v = base[i]; s += v; s2 += v * v; }
    __shared__ float rs[256], rq[256];
    rs[tid] = s; rq[tid] = s2; __syncthreads();
    for (int off = 128; off > 0; off >>= 1) {
        if (tid < off) { rs[tid] += rs[tid + off]; rq[tid] += rq[tid + off]; }
        __syncthreads();
    }
    float mean = rs[0] / n;
    float var  = rq[0] / n - mean * mean;
    float inv  = rsqrtf(var + 1e-5f);
    for (int i = tid; i < n; i += 256) {
        int c = g * 5 + i / (SS * TP);
        float v = (base[i] - mean) * inv * gs[c] + gb[c];
        base[i] = 0.5f * v * (1.f + erff(v * 0.70710678118654752f));
    }
}

// ---------------------------------------------------------------------------
// conv with kernel (1,3), pad 1, 25->25 channels, in/out [B,25,570,8]
// ---------------------------------------------------------------------------
__global__ __launch_bounds__(256) void k_conv3(const float* __restrict__ in,
                                               const float* __restrict__ w,
                                               const float* __restrict__ bias,
                                               float* __restrict__ out) {
    long i = (long)blockIdx.x * 256 + threadIdx.x;
    if (i >= (long)BB * OC * SS * TP) return;
    int tp = i & 7;
    long r  = i >> 3;
    int s  = r % SS; long r2 = r / SS;
    int oc = r2 % OC; int b = r2 / OC;
    const float* wr = w + oc * 75;
    float acc = bias[oc];
#pragma unroll 5
    for (int ic = 0; ic < OC; ic++) {
        const float* row = in + (((size_t)b * OC + ic) * SS + s) * TP;
#pragma unroll
        for (int k = 0; k < 3; k++) {
            int t = tp + k - 1;
            if (t >= 0 && t < TP) acc += wr[ic * 3 + k] * row[t];
        }
    }
    out[i] = acc;
}

// ---------------------------------------------------------------------------
// h3[B,25,570,8] -> patch_emb[TT, 200]  (transpose(0,2,1,3).reshape)
// ---------------------------------------------------------------------------
__global__ __launch_bounds__(256) void k_assemble(const float* __restrict__ h,
                                                  float* __restrict__ pemb) {
    long i = (long)blockIdx.x * 256 + threadIdx.x;
    if (i >= (long)TT * DD) return;
    int dd = i % DD; long tok = i / DD;
    int oc = dd >> 3, tp = dd & 7;
    int b = tok / SS, s = tok % SS;
    pemb[i] = h[(((size_t)b * OC + oc) * SS + s) * TP + tp];
}

// ---------------------------------------------------------------------------
// twiddle table with (f*t) mod 200 argument reduction
// ---------------------------------------------------------------------------
__global__ void k_twiddle(float* __restrict__ twc, float* __restrict__ tws) {
    int i = blockIdx.x * 256 + threadIdx.x;
    if (i >= NFREQ * TL) return;
    int f = i / TL, t = i % TL;
    int m = (f * t) % TL;
    float ang = -6.28318530717958647692f * (float)m / (float)TL;
    float sv, cv; sincosf(ang, &sv, &cv);
    twc[i] = cv; tws[i] = sv;
}

// |rfft(x, norm="forward")| -> amp bf16 [TT, 128] (freq 101..127 zero)
__global__ __launch_bounds__(128) void k_dft(const float* __restrict__ x,
                                             const float* __restrict__ twc,
                                             const float* __restrict__ tws,
                                             bf16_t* __restrict__ amp) {
    int row = blockIdx.x;
    __shared__ float xr[TL];
    for (int t = threadIdx.x; t < TL; t += 128) xr[t] = x[(size_t)row * TL + t];
    __syncthreads();
    int f = threadIdx.x;
    if (f < NFREQ) {
        const float* tc = twc + f * TL;
        const float* ts = tws + f * TL;
        float cr = 0.f, ci = 0.f;
        for (int t = 0; t < TL; t++) { float v = xr[t]; cr += v * tc[t]; ci += v * ts[t]; }
        amp[(size_t)row * KF + f] = (bf16_t)(sqrtf(cr * cr + ci * ci) * (1.f / TL));
    } else {
        amp[(size_t)row * KF + f] = (bf16_t)0.f;
    }
}

// ---------------------------------------------------------------------------
// transpose + pad weights to bf16 column-major Bt[Npad][Kpad]; src [R][Ccols]
// grid.y = matrix index
// ---------------------------------------------------------------------------
__global__ void k_tpad(const float* __restrict__ src, bf16_t* __restrict__ dst,
                       int R, int Ccols, int Kpad, int Npad) {
    long i = (long)blockIdx.x * 256 + threadIdx.x;
    if (i >= (long)Npad * Kpad) return;
    int k = i % Kpad, n = i / Kpad;
    const float* s = src + (size_t)blockIdx.y * R * Ccols;
    bf16_t* d = dst + (size_t)blockIdx.y * Npad * Kpad;
    float v = (n < Ccols && k < R) ? s[(size_t)k * Ccols + n] : 0.f;
    d[i] = (bf16_t)v;
}

// ---------------------------------------------------------------------------
// helpers for WMMA fragments
// ---------------------------------------------------------------------------
__device__ __forceinline__ v16bf load_a_frag(const bf16_t* p) {
    // p = row_base + khalf*8 + k0 ; comps 0..7 <- p[0..7], 8..15 <- p[16..23]
    v8bf lo = *(const v8bf*)(p);
    v8bf hi = *(const v8bf*)(p + 16);
    v16bf a;
#pragma unroll
    for (int q = 0; q < 8; q++) { a[q] = lo[q]; a[q + 8] = hi[q]; }
    return a;
}
__device__ __forceinline__ v16bf load_b_frag(const bf16_t* p) {
    // p = col_base + khalf*8 + k0 ; comps 0..7 <- p[0..7], 8..15 <- p[16..23] (two 16B LDS loads)
    v8bf lo = *(const v8bf*)(p);
    v8bf hi = *(const v8bf*)(p + 16);
    v16bf b;
#pragma unroll
    for (int q = 0; q < 8; q++) { b[q] = lo[q]; b[q + 8] = hi[q]; }
    return b;
}

// ---------------------------------------------------------------------------
// WMMA GEMM: 128-thread block = 4 waves x 32-row subtiles sharing one 32-col
// N-tile whose B strip [32 x K] is staged in LDS (async global->LDS via the
// CDNA5 ASYNCcnt builtins when available; plain ds stores otherwise).
// Per wave: 32x32 register tile, 4 WMMA per K-step (2 A frags x 2 B frags).
// C[M,N] (f32 acc) = A[M,K](bf16 row-major) x Bt[Npad,K](bf16 col-major) +bias
// MODE 0: Cf += (read existing Cf, add bias)   [spectral]
// MODE 1: Cb = bf16(gelu(. + bias))            [FFN1]
// MODE 2: Cf = . + bias                        [FFN2]
// KROW = K + 8 (LDS row padding: bank-conflict-free 16-lane column access)
// M mult of 128; Npad, K mult of 32; blockIdx.y = matrix (expert)
// ---------------------------------------------------------------------------
template <int MODE, int KROW>
__global__ __launch_bounds__(128) void k_gemm(const bf16_t* __restrict__ A,
                                              const bf16_t* __restrict__ Bt,
                                              const float* __restrict__ bias,
                                              float* __restrict__ Cf,
                                              bf16_t* __restrict__ Cb,
                                              int M, int N, int K,
                                              long sA, long sB, long sBias, long sC) {
    __shared__ bf16_t ldsB[32 * KROW];

    const int mat  = blockIdx.y;
    const bf16_t* Am = A  + (size_t)mat * sA;
    const bf16_t* Bm = Bt + (size_t)mat * sB;
    const float*  bm = bias + (size_t)mat * sBias;
    float*  Cfm = Cf ? Cf + (size_t)mat * sC : nullptr;
    bf16_t* Cbm = Cb ? Cb + (size_t)mat * sC : nullptr;

    const int ntiles = (N + 31) >> 5;
    const int mb = blockIdx.x / ntiles;
    const int nt = blockIdx.x % ntiles;
    const int lane = threadIdx.x & 31;
    const int wave = threadIdx.x >> 5;
    const int m0 = mb * 128 + wave * 32;
    const int n0 = nt * 32;

    // ---- stage B strip rows [n0, n0+32) into LDS (row stride KROW elems) ----
    {
        const int cpr = K >> 3; // 16-byte chunks per row
        const bf16_t* gB = Bm + (size_t)n0 * K;
        for (int ch = threadIdx.x; ch < 32 * cpr; ch += 128) {
            int rowi = ch / cpr, ci = ch - rowi * cpr;
            const bf16_t* src = gB + (size_t)rowi * K + ci * 8;
            bf16_t* dst = &ldsB[rowi * KROW + ci * 8];
#if GEMM_ASYNC
            __builtin_amdgcn_global_load_async_to_lds_b128(
                (v4i_vs*)src, (lds_v4i_p)dst, 0, 0);
#else
            *(v8bf*)dst = *(const v8bf*)src;
#endif
        }
#if GEMM_ASYNC
        __builtin_amdgcn_s_wait_asynccnt(0);
#endif
        __syncthreads();
    }

    const int l15   = lane & 15;
    const int khalf = lane >> 4;

    // accumulators c[mi][nj]
    v8f c00, c01, c10, c11;
    const int ncol0 = n0 + l15, ncol1 = n0 + 16 + l15;
    float b0i = (ncol0 < N) ? bm[ncol0] : 0.f;
    float b1i = (ncol1 < N) ? bm[ncol1] : 0.f;
#pragma unroll
    for (int r = 0; r < 8; r++) {
        int mA = m0 + r + (khalf << 3);
        int mB = mA + 16;
        float i00 = b0i, i01 = b1i, i10 = b0i, i11 = b1i;
        if (MODE == 0) {
            i00 += (ncol0 < N) ? Cfm[(size_t)mA * N + ncol0] : 0.f;
            i01 += (ncol1 < N) ? Cfm[(size_t)mA * N + ncol1] : 0.f;
            i10 += (ncol0 < N) ? Cfm[(size_t)mB * N + ncol0] : 0.f;
            i11 += (ncol1 < N) ? Cfm[(size_t)mB * N + ncol1] : 0.f;
        }
        c00[r] = i00; c01[r] = i01; c10[r] = i10; c11[r] = i11;
    }

    const bf16_t* aBase0 = Am + (size_t)(m0 + l15) * K + khalf * 8;
    const bf16_t* aBase1 = aBase0 + (size_t)16 * K;
    const bf16_t* lB0 = &ldsB[(size_t)l15 * KROW + khalf * 8];
    const bf16_t* lB1 = lB0 + (size_t)16 * KROW;

    for (int k0 = 0; k0 < K; k0 += 32) {
        v16bf a0 = load_a_frag(aBase0 + k0);
        v16bf a1 = load_a_frag(aBase1 + k0);
        v16bf b0 = load_b_frag(lB0 + k0);
        v16bf b1 = load_b_frag(lB1 + k0);
        c00 = __builtin_amdgcn_wmma_f32_16x16x32_bf16(false, a0, false, b0, (short)0, c00, false, false);
        c01 = __builtin_amdgcn_wmma_f32_16x16x32_bf16(false, a0, false, b1, (short)0, c01, false, false);
        c10 = __builtin_amdgcn_wmma_f32_16x16x32_bf16(false, a1, false, b0, (short)0, c10, false, false);
        c11 = __builtin_amdgcn_wmma_f32_16x16x32_bf16(false, a1, false, b1, (short)0, c11, false, false);
    }

#pragma unroll
    for (int r = 0; r < 8; r++) {
        int mA = m0 + r + (khalf << 3);
        int mB = mA + 16;
        float v00 = c00[r], v01 = c01[r], v10 = c10[r], v11 = c11[r];
        if (MODE == 1) {
            v00 = 0.5f * v00 * (1.f + erff(v00 * 0.70710678118654752f));
            v01 = 0.5f * v01 * (1.f + erff(v01 * 0.70710678118654752f));
            v10 = 0.5f * v10 * (1.f + erff(v10 * 0.70710678118654752f));
            v11 = 0.5f * v11 * (1.f + erff(v11 * 0.70710678118654752f));
            if (ncol0 < N) { Cbm[(size_t)mA * N + ncol0] = (bf16_t)v00;
                             Cbm[(size_t)mB * N + ncol0] = (bf16_t)v10; }
            if (ncol1 < N) { Cbm[(size_t)mA * N + ncol1] = (bf16_t)v01;
                             Cbm[(size_t)mB * N + ncol1] = (bf16_t)v11; }
        } else {
            if (ncol0 < N) { Cfm[(size_t)mA * N + ncol0] = v00;
                             Cfm[(size_t)mB * N + ncol0] = v10; }
            if (ncol1 < N) { Cfm[(size_t)mA * N + ncol1] = v01;
                             Cfm[(size_t)mB * N + ncol1] = v11; }
        }
    }
}

// ---------------------------------------------------------------------------
// depthwise positional conv (kernel 19x7 over (C,P), pad (9,3)) + residual
// ---------------------------------------------------------------------------
__global__ __launch_bounds__(256) void k_posconv(const float* __restrict__ pemb,
                                                 const float* __restrict__ pw,
                                                 const float* __restrict__ pb,
                                                 float* __restrict__ pe) {
    long i = (long)blockIdx.x * 256 + threadIdx.x;
    if (i >= (long)TT * DD) return;
    int dd = i % DD; long tok = i / DD;
    int b = tok / SS, sp = tok % SS;
    int c = sp / PP, p = sp % PP;
    float acc = pemb[i] + pb[dd];
    const float* wk = pw + (size_t)dd * 133;
    const float* pbase = pemb + (size_t)b * SS * DD + dd;
    for (int kc = 0; kc < 19; kc++) {
        int c2 = c + kc - 9;
        if (c2 < 0 || c2 >= CC) continue;
#pragma unroll
        for (int kp = 0; kp < 7; kp++) {
            int p2 = p + kp - 3;
            if (p2 < 0 || p2 >= PP) continue;
            acc += wk[kc * 7 + kp] * pbase[(size_t)(c2 * PP + p2) * DD];
        }
    }
    pe[i] = acc;
}

// ---------------------------------------------------------------------------
// router: logits = pe @ router_w + b; expert = first-argmax; gate = max softmax
// ---------------------------------------------------------------------------
__global__ __launch_bounds__(256) void k_router(const float* __restrict__ pe,
                                                const float* __restrict__ rw,
                                                const float* __restrict__ rb,
                                                int* __restrict__ expert,
                                                float* __restrict__ gate) {
    int tok = blockIdx.x * 256 + threadIdx.x;
    if (tok >= TT) return;
    float l0 = rb[0], l1 = rb[1], l2 = rb[2], l3 = rb[3];
    const float* tr = pe + (size_t)tok * DD;
    for (int dd = 0; dd < DD; dd++) {
        float v = tr[dd];
        const float* r = rw + dd * 4;
        l0 += v * r[0]; l1 += v * r[1]; l2 += v * r[2]; l3 += v * r[3];
    }
    float m = l0; int e = 0;
    if (l1 > m) { m = l1; e = 1; }
    if (l2 > m) { m = l2; e = 2; }
    if (l3 > m) { m = l3; e = 3; }
    float s = expf(l0 - m) + expf(l1 - m) + expf(l2 - m) + expf(l3 - m);
    expert[tok] = e;
    gate[tok] = 1.f / s;
}

// ---------------------------------------------------------------------------
// deterministic ordered dispatch (matches jnp.cumsum token order exactly)
// ---------------------------------------------------------------------------
__global__ __launch_bounds__(256) void k_scan_local(const int* __restrict__ expert,
                                                    int* __restrict__ lrank,
                                                    int* __restrict__ blkcnt) {
    int tid = threadIdx.x;
    int tok = blockIdx.x * 256 + tid;
    int e = expert[tok];
    __shared__ unsigned sA[256], sB[256];
    sA[tid] = (e == 0 ? 1u : 0u) | ((e == 1 ? 1u : 0u) << 16);
    sB[tid] = (e == 2 ? 1u : 0u) | ((e == 3 ? 1u : 0u) << 16);
    __syncthreads();
    for (int off = 1; off < 256; off <<= 1) {
        unsigned na = 0, nb = 0;
        if (tid >= off) { na = sA[tid - off]; nb = sB[tid - off]; }
        __syncthreads();
        sA[tid] += na; sB[tid] += nb;
        __syncthreads();
    }
    unsigned ia = sA[tid], ib = sB[tid];
    int incl = (e == 0) ? (ia & 0xffff) : (e == 1) ? (ia >> 16)
             : (e == 2) ? (ib & 0xffff) : (ib >> 16);
    lrank[tok] = incl - 1;
    if (tid == 255) {
        blkcnt[blockIdx.x * 4 + 0] = sA[255] & 0xffff;
        blkcnt[blockIdx.x * 4 + 1] = sA[255] >> 16;
        blkcnt[blockIdx.x * 4 + 2] = sB[255] & 0xffff;
        blkcnt[blockIdx.x * 4 + 3] = sB[255] >> 16;
    }
}

__global__ void k_scan_blocks(const int* __restrict__ blkcnt,
                              int* __restrict__ blkbase, int nblk) {
    int e = threadIdx.x;
    if (e < 4) {
        int run = 0;
        for (int b = 0; b < nblk; b++) { blkbase[b * 4 + e] = run; run += blkcnt[b * 4 + e]; }
    }
}

__global__ __launch_bounds__(256) void k_dispatch(const int* __restrict__ expert,
                                                  const int* __restrict__ lrank,
                                                  const int* __restrict__ blkbase,
                                                  const float* __restrict__ gate,
                                                  int* __restrict__ idxb,
                                                  float* __restrict__ gk) {
    int tok = blockIdx.x * 256 + threadIdx.x;
    if (tok >= TT) return;
    int e = expert[tok];
    int pos = blkbase[(tok >> 8) * 4 + e] + lrank[tok];
    bool keep = pos < CAP;
    idxb[tok] = keep ? e * CAP + pos : -1;
    gk[tok] = keep ? gate[tok] : 0.f;
}

// scatter kept tokens into bf16 dispatch buffer [TT, KD] (pre-zeroed)
__global__ __launch_bounds__(256) void k_scatter(const float* __restrict__ pe,
                                                 const int* __restrict__ idxb,
                                                 bf16_t* __restrict__ buf) {
    int tok = blockIdx.x, dd = threadIdx.x;
    int slot = idxb[tok];
    if (slot < 0 || dd >= DD) return;
    buf[(size_t)slot * KD + dd] = (bf16_t)pe[(size_t)tok * DD + dd];
}

// out = pe + gate*keep * y[slot]
__global__ __launch_bounds__(256) void k_gather(const float* __restrict__ pe,
                                                const int* __restrict__ idxb,
                                                const float* __restrict__ gk,
                                                const float* __restrict__ y,
                                                float* __restrict__ out) {
    long i = (long)blockIdx.x * 256 + threadIdx.x;
    if (i >= (long)TT * DD) return;
    int dd = i % DD; long tok = i / DD;
    int s = idxb[tok];
    float add = (s >= 0) ? gk[tok] * y[(size_t)s * DD + dd] : 0.f;
    out[i] = pe[i] + add;
}

// ---------------------------------------------------------------------------
extern "C" void kernel_launch(void* const* d_in, const int* in_sizes, int n_in,
                              void* d_out, int out_size, void* d_ws, size_t ws_size,
                              hipStream_t stream) {
    const float* x        = (const float*)d_in[0];
    const float* conv1_w  = (const float*)d_in[1];
    const float* conv1_b  = (const float*)d_in[2];
    const float* gn1_s    = (const float*)d_in[3];
    const float* gn1_b    = (const float*)d_in[4];
    const float* conv2_w  = (const float*)d_in[5];
    const float* conv2_b  = (const float*)d_in[6];
    const float* gn2_s    = (const float*)d_in[7];
    const float* gn2_b    = (const float*)d_in[8];
    const float* conv3_w  = (const float*)d_in[9];
    const float* conv3_b  = (const float*)d_in[10];
    const float* gn3_s    = (const float*)d_in[11];
    const float* gn3_b    = (const float*)d_in[12];
    const float* spec_w   = (const float*)d_in[13];
    const float* spec_b   = (const float*)d_in[14];
    const float* pos_w    = (const float*)d_in[15];
    const float* pos_b    = (const float*)d_in[16];
    const float* router_w = (const float*)d_in[17];
    const float* router_b = (const float*)d_in[18];
    const float* e_w1     = (const float*)d_in[19];
    const float* e_b1     = (const float*)d_in[20];
    const float* e_w2     = (const float*)d_in[21];
    const float* e_b2     = (const float*)d_in[22];
    float* out = (float*)d_out;

    char* ws = (char*)d_ws;
    float*  hA    = (float*)(ws + OFF_HA);
    float*  hB    = (float*)(ws + OFF_HB);
    float*  pemb  = (float*)(ws + OFF_PEMB);
    bf16_t* amp   = (bf16_t*)(ws + OFF_AMP);
    float*  twc   = (float*)(ws + OFF_TW);
    float*  tws   = twc + (size_t)NFREQ * TL;
    bf16_t* wspec = (bf16_t*)(ws + OFF_WSPEC);
    bf16_t* w1t   = (bf16_t*)(ws + OFF_W1T);
    bf16_t* w2t   = (bf16_t*)(ws + OFF_W2T);
    int*    expr  = (int*)(ws + OFF_EXP);
    float*  gate  = (float*)(ws + OFF_GATE);
    int*    lrank = (int*)(ws + OFF_LRNK);
    int*    bcnt  = (int*)(ws + OFF_BCNT);
    int*    bbas  = (int*)(ws + OFF_BBAS);
    int*    idxb  = (int*)(ws + OFF_IDX);
    float*  gk    = (float*)(ws + OFF_GK);
    bf16_t* hid   = (bf16_t*)(ws + OFF_HID);
    bf16_t* buf   = (bf16_t*)(ws + OFF_BUF);
    float*  yv    = (float*)(ws + OFF_Y);
    float*  pe    = (float*)(ws + OFF_PE);

    const long NELEM_H = (long)BB * OC * SS * TP; // 29,184,000
    const long NELEM_P = (long)TT * DD;           // 29,184,000

    // weight prep (bf16, transposed, padded)
    k_twiddle<<<(NFREQ * TL + 255) / 256, 256, 0, stream>>>(twc, tws);
    k_tpad<<<dim3((NP * KF + 255) / 256, 1), 256, 0, stream>>>(spec_w, wspec, NFREQ, DD, KF, NP);
    k_tpad<<<dim3((DFF * KD + 255) / 256, NE), 256, 0, stream>>>(e_w1, w1t, DD, DFF, KD, DFF);
    k_tpad<<<dim3((NP * DFF + 255) / 256, NE), 256, 0, stream>>>(e_w2, w2t, DFF, DD, DFF, NP);

    // temporal conv stack with fused GN+GELU
    k_conv1<<<(NELEM_H + 255) / 256, 256, 0, stream>>>(x, conv1_w, conv1_b, hA);
    k_gn_gelu<<<BB * 5, 256, 0, stream>>>(hA, gn1_s, gn1_b);
    k_conv3<<<(NELEM_H + 255) / 256, 256, 0, stream>>>(hA, conv2_w, conv2_b, hB);
    k_gn_gelu<<<BB * 5, 256, 0, stream>>>(hB, gn2_s, gn2_b);
    k_conv3<<<(NELEM_H + 255) / 256, 256, 0, stream>>>(hB, conv3_w, conv3_b, hA);
    k_gn_gelu<<<BB * 5, 256, 0, stream>>>(hA, gn3_s, gn3_b);
    k_assemble<<<(NELEM_P + 255) / 256, 256, 0, stream>>>(hA, pemb);

    // spectral branch: DFT magnitudes then WMMA GEMM accumulate into pemb
    k_dft<<<TT, 128, 0, stream>>>(x, twc, tws, amp);
    k_gemm<0, KF + 8><<<dim3((TT / 128) * ((DD + 31) / 32), 1), 128, 0, stream>>>(
        amp, wspec, spec_b, pemb, nullptr, TT, DD, KF, 0, 0, 0, 0);

    // depthwise positional conv + residual -> pe
    k_posconv<<<(NELEM_P + 255) / 256, 256, 0, stream>>>(pemb, pos_w, pos_b, pe);

    // MoE: router -> deterministic ordered dispatch
    k_router<<<(TT + 255) / 256, 256, 0, stream>>>(pe, router_w, router_b, expr, gate);
    k_scan_local<<<TT / 256, 256, 0, stream>>>(expr, lrank, bcnt);
    k_scan_blocks<<<1, 32, 0, stream>>>(bcnt, bbas, TT / 256);
    k_dispatch<<<(TT + 255) / 256, 256, 0, stream>>>(expr, lrank, bbas, gate, idxb, gk);
    (void)hipMemsetAsync(buf, 0, (size_t)TT * KD * sizeof(bf16_t), stream);
    k_scatter<<<TT, 256, 0, stream>>>(pe, idxb, buf);

    // expert FFN: bf16 WMMA GEMMs
    k_gemm<1, KD + 8><<<dim3((CAP / 128) * (DFF / 32), NE), 128, 0, stream>>>(
        buf, w1t, e_b1, nullptr, hid, CAP, DFF, KD,
        (long)CAP * KD, (long)DFF * KD, DFF, (long)CAP * DFF);
    k_gemm<2, DFF + 8><<<dim3((CAP / 128) * ((DD + 31) / 32), NE), 128, 0, stream>>>(
        hid, w2t, e_b2, yv, nullptr, CAP, DD, DFF,
        (long)CAP * DFF, (long)NP * DFF, DD, (long)CAP * DD);

    // combine: out = pe + gate*keep * expert_out
    k_gather<<<(NELEM_P + 255) / 256, 256, 0, stream>>>(pe, idxb, gk, yv, out);

    (void)in_sizes; (void)n_in; (void)out_size; (void)ws_size;
}
